// ConditionalStrNN_58772332478466
// MI455X (gfx1250) — matrix operand order repaired and checked by reading, not compiled, not run on hardware
//
#include <hip/hip_runtime.h>

// ---------------------------------------------------------------------------
// ConditionalStrNN (MADE-style) for MI455X / gfx1250 (wave32, WMMA).
//
// Problem sizes: B=64, NIN=512, H1=H2=1024, NOUT=512, COND=64, FC=1.
//
// Per layer:
//   f1[b,i] = u @ aw.T + ab           (64 x I   GEMM, K=64)  -> WMMA f16
//   f2[b,o] = u @ bw.T + bb           (64 x O   GEMM, K=64)  -> WMMA f16
//   out[b,o] = sum_i sigmoid(f1[b,i]*f2[b,o]) * mask[o,i]*W[o,i] * x[b,i]
//
// The masked contraction has a batch-dependent effective weight matrix
// (sigmoid of a rank-1 term), so it cannot be a GEMM; it is VALU/TRANS
// bound (~100M sigmoids). Weights/masks (~16MB) are L2 resident; each
// mask*W row is staged once in LDS and reused by all 64 batches, and each
// x/f1 load is reused across an O-tile of 4 neurons (4x less L2 traffic).
// ---------------------------------------------------------------------------

typedef _Float16 v16h __attribute__((ext_vector_type(16)));
typedef float    v8f  __attribute__((ext_vector_type(8)));

#define BCOUNT 64
#define CONDK  64

// ---------------------------------------------------------------------------
// WMMA conditioner GEMM:  out[m,n] = sum_k u[m,k]*w[n,k] + bias[n]
// M = 64 (batch), K = 64 (COND), N = feature count (512 or 1024).
// Block = 128 threads = 4 waves; wave w owns M-tile [16w,16w+16);
// blockIdx.x owns N-tile. K=64 -> two v_wmma_f32_16x16x32_f16 steps.
//
// Lane layouts (ISA 7.12.2, wave32):
//   A 16x32 f16 : lane l -> M = l&15 ; hi = l>>4 ;
//                 half h: K = (h<8) ? 8*hi+h : 16+8*hi+(h-8)
//   B 32x16 f16 : lane l -> N = l&15 ; half h: K = (h<8) ? 8*hi+h : 16+8*hi+(h-8)
//   D 16x16 f32 : lane l -> N = l&15 ; VGPR v: M = v + 8*hi
// ---------------------------------------------------------------------------
__global__ __launch_bounds__(128)
void cond_gemm_wmma(const float* __restrict__ u,      // (64, 64)
                    const float* __restrict__ w,      // (N, 64)
                    const float* __restrict__ bias,   // (N,)
                    float* __restrict__ out,          // (64, N)
                    int N) {
  const int wave = threadIdx.x >> 5;
  const int lane = threadIdx.x & 31;
  const int lo   = lane & 15;
  const int hi   = lane >> 4;
  const int m0   = wave * 16;
  const int n0   = blockIdx.x * 16;

  v8f c = {};
#pragma unroll
  for (int kstep = 0; kstep < 2; ++kstep) {
    const int kb = kstep * 32;
    v16h a, b;
#pragma unroll
    for (int h = 0; h < 16; ++h) {
      const int ka = kb + ((h < 8) ? (8 * hi + h) : (16 + 8 * hi + (h - 8)));
      a[h] = (_Float16)u[(m0 + lo) * CONDK + ka];
      const int kbm = kb + ((h < 8) ? (8 * hi + h) : (16 + 8 * hi + (h - 8)));
      b[h] = (_Float16)w[(n0 + lo) * CONDK + kbm];
    }
    // 8 args: (neg_a, A, neg_b, B, c_mod, C, reuse_a, reuse_b)
    c = __builtin_amdgcn_wmma_f32_16x16x32_f16(false, a, false, b,
                                               (short)0, c, false, false);
  }

  const int   n  = n0 + lo;
  const float bv = bias[n];
#pragma unroll
  for (int v = 0; v < 8; ++v) {
    const int m = m0 + v + 8 * hi;
    out[m * N + n] = c[v] + bv;
  }
}

// ---------------------------------------------------------------------------
// Fused masked layer: one block (256 thr = 8 waves) per O-tile of OT=4
// output neurons.
//   - stage mw[ot][i] = mask[o,i]*W[o,i] in LDS (reused by all 64 batches)
//   - wave v handles batches b = v, v+8, ...; lanes stride the I dimension
//     with float4 (global_load_b128) loads, each reused across the OT rows
//   - sigmoid via v_exp_f32/v_rcp_f32; -log2(e) pre-folded into f2 so the
//     per-element chain is: mul, exp, add, rcp, mul, fma
//   - wave32 __shfl_xor butterfly reduction, lane 0 stores out[b,o]
// ---------------------------------------------------------------------------
template <int I, int O, int OT>
__global__ __launch_bounds__(256)
void cml_layer(const float* __restrict__ x,     // (64, I)
               const float* __restrict__ f1,    // (64, I)
               const float* __restrict__ f2,    // (64, O)
               const float* __restrict__ mask,  // (O, I)
               const float* __restrict__ W,     // (O, I)
               float* __restrict__ out) {       // (64, O)
  __shared__ __align__(16) float mw[OT * I];
  const int o0  = blockIdx.x * OT;
  const int tid = threadIdx.x;

  // Stage mask*W rows for this O-tile (float4 granularity).
  for (int idx = tid; idx < OT * I / 4; idx += 256) {
    const int ot = idx / (I / 4);
    const int ii = (idx % (I / 4)) * 4;
    const float4 m4 = *(const float4*)(mask + (o0 + ot) * I + ii);
    const float4 w4 = *(const float4*)(W    + (o0 + ot) * I + ii);
    float4 p;
    p.x = m4.x * w4.x; p.y = m4.y * w4.y; p.z = m4.z * w4.z; p.w = m4.w * w4.w;
    *(float4*)(mw + ot * I + ii) = p;
  }
  __syncthreads();

  const int   wave   = tid >> 5;
  const int   lane   = tid & 31;
  const float NLOG2E = -1.44269504088896340736f;

  for (int b = wave; b < BCOUNT; b += 8) {
    float f2s[OT];
    float acc[OT];
#pragma unroll
    for (int ot = 0; ot < OT; ++ot) {
      f2s[ot] = f2[b * O + o0 + ot] * NLOG2E;  // fold -log2e once per (b,o)
      acc[ot] = 0.0f;
    }

    for (int i0 = lane * 4; i0 < I; i0 += 128) {
      const float4 f1v = *(const float4*)(f1 + b * I + i0);
      const float4 xv  = *(const float4*)(x  + b * I + i0);
#pragma unroll
      for (int ot = 0; ot < OT; ++ot) {
        const float4 mv = *(const float4*)(mw + ot * I + i0);
        // sigmoid(f1*f2) = rcp(1 + exp2(f1 * (f2 * -log2e)))
        {
          const float s = __builtin_amdgcn_rcpf(
              1.0f + __builtin_amdgcn_exp2f(f1v.x * f2s[ot]));
          acc[ot] += s * mv.x * xv.x;
        }
        {
          const float s = __builtin_amdgcn_rcpf(
              1.0f + __builtin_amdgcn_exp2f(f1v.y * f2s[ot]));
          acc[ot] += s * mv.y * xv.y;
        }
        {
          const float s = __builtin_amdgcn_rcpf(
              1.0f + __builtin_amdgcn_exp2f(f1v.z * f2s[ot]));
          acc[ot] += s * mv.z * xv.z;
        }
        {
          const float s = __builtin_amdgcn_rcpf(
              1.0f + __builtin_amdgcn_exp2f(f1v.w * f2s[ot]));
          acc[ot] += s * mv.w * xv.w;
        }
      }
    }

#pragma unroll
    for (int ot = 0; ot < OT; ++ot) {
      float a = acc[ot];
#pragma unroll
      for (int off = 16; off > 0; off >>= 1) {
        a += __shfl_xor(a, off, 32);
      }
      if (lane == 0) out[b * O + o0 + ot] = a;
    }
  }
}

// ---------------------------------------------------------------------------
// Launch: 6 WMMA conditioner GEMMs, then 3 fused masked layers (chained
// through d_ws). All on `stream`, fully deterministic, no allocation.
// ---------------------------------------------------------------------------
extern "C" void kernel_launch(void* const* d_in, const int* in_sizes, int n_in,
                              void* d_out, int out_size, void* d_ws, size_t ws_size,
                              hipStream_t stream) {
  (void)in_sizes; (void)n_in; (void)out_size; (void)ws_size;

  const int NIN = 512, H1 = 1024, H2 = 1024, NOUT = 512;

  const float* x     = (const float*)d_in[0];
  const float* u     = (const float*)d_in[1];
  const float* mask0 = (const float*)d_in[2];
  const float* mask1 = (const float*)d_in[3];
  const float* mask2 = (const float*)d_in[4];
  const float* W0    = (const float*)d_in[5];
  const float* W1    = (const float*)d_in[6];
  const float* W2    = (const float*)d_in[7];
  const float* a0_w  = (const float*)d_in[8];
  const float* a0_b  = (const float*)d_in[9];
  const float* b0_w  = (const float*)d_in[10];
  const float* b0_b  = (const float*)d_in[11];
  const float* a1_w  = (const float*)d_in[12];
  const float* a1_b  = (const float*)d_in[13];
  const float* b1_w  = (const float*)d_in[14];
  const float* b1_b  = (const float*)d_in[15];
  const float* a2_w  = (const float*)d_in[16];
  const float* a2_b  = (const float*)d_in[17];
  const float* b2_w  = (const float*)d_in[18];
  const float* b2_b  = (const float*)d_in[19];

  float* ws = (float*)d_ws;
  // Workspace layout (floats):
  float* f1_0 = ws;                         // 64*512  = 32768
  float* f2_0 = f1_0 + 64 * NIN;            // 64*1024 = 65536
  float* f1_1 = f2_0 + 64 * H1;             // 64*1024
  float* f2_1 = f1_1 + 64 * H1;             // 64*1024
  float* f1_2 = f2_1 + 64 * H2;             // 64*1024
  float* f2_2 = f1_2 + 64 * H2;             // 64*512
  float* h1   = f2_2 + 64 * NOUT;           // 64*1024
  float* h2   = h1   + 64 * H1;             // 64*1024  (end ~1.75 MB)

  dim3 gblk(128);
  // Conditioner GEMMs (WMMA):  f = u @ w.T + bias
  cond_gemm_wmma<<<dim3(NIN  / 16), gblk, 0, stream>>>(u, a0_w, a0_b, f1_0, NIN);
  cond_gemm_wmma<<<dim3(H1   / 16), gblk, 0, stream>>>(u, b0_w, b0_b, f2_0, H1);
  cond_gemm_wmma<<<dim3(H1   / 16), gblk, 0, stream>>>(u, a1_w, a1_b, f1_1, H1);
  cond_gemm_wmma<<<dim3(H2   / 16), gblk, 0, stream>>>(u, b1_w, b1_b, f2_1, H2);
  cond_gemm_wmma<<<dim3(H2   / 16), gblk, 0, stream>>>(u, a2_w, a2_b, f1_2, H2);
  cond_gemm_wmma<<<dim3(NOUT / 16), gblk, 0, stream>>>(u, b2_w, b2_b, f2_2, NOUT);

  // Masked layers (VALU/TRANS bound; one block per 4 output neurons)
  cml_layer<512, 1024, 4><<<dim3(1024 / 4), dim3(256), 0, stream>>>(x,  f1_0, f2_0, mask0, W0, h1);
  cml_layer<1024, 1024, 4><<<dim3(1024 / 4), dim3(256), 0, stream>>>(h1, f1_1, f2_1, mask1, W1, h2);
  cml_layer<1024, 512, 4><<<dim3(512 / 4),  dim3(256), 0, stream>>>(h2, f1_2, f2_2, mask2, W2, (float*)d_out);
}
